// _VisionAttention_3461743640979
// MI455X (gfx1250) — matrix operand order repaired and checked by reading, not compiled, not run on hardware
//
#include <hip/hip_runtime.h>

typedef __attribute__((ext_vector_type(16))) _Float16 v16h;
typedef __attribute__((ext_vector_type(8)))  float    v8f;
typedef __attribute__((ext_vector_type(4)))  int      v4i;

namespace {
constexpr int BB  = 2;
constexpr int SS  = 2048;
constexpr int HID = 1024;
constexpr int NH  = 16;
constexpr int NKV = 8;
constexpr int DD  = 64;
constexpr int NQKV = NH * DD + 2 * NKV * DD;   // 2048
constexpr float EPS = 1e-6f;
}

#if defined(__has_builtin)
#if __has_builtin(__builtin_amdgcn_global_load_async_to_lds_b128) && \
    __has_builtin(__builtin_amdgcn_s_wait_asynccnt)
#define USE_ASYNC 1
#endif
#endif

union FragU { v16h v; unsigned int u[8]; };

// A-matrix 16x32 f16 fragment (ISA 7.12.2): lane m=L%16, half=L/16,
// VGPR r: k = (r>>2)*16 + half*8 + 2*(r&3), pair {k,k+1}. Rows contiguous in memory.
static __device__ __forceinline__ v16h load_fragA(const _Float16* base, int ld,
                                                  int row0, int k0, int lane) {
  const int m = lane & 15, half = lane >> 4;
  FragU f;
#pragma unroll
  for (int r = 0; r < 8; ++r) {
    const int k = k0 + ((r >> 2) * 16) + half * 8 + 2 * (r & 3);
    f.u[r] = *(const unsigned int*)(base + (size_t)(row0 + m) * ld + k);
  }
  return f.v;
}

// B-matrix 32x16 f16 fragment: lane n=L%16 holds column n, VGPR r: k = (r>>2)*16 + half*8 + 2*(r&3).
// Memory holds column n contiguously over k at base[(col0+n)*ld + k].
static __device__ __forceinline__ v16h load_fragB(const _Float16* base, int ld,
                                                  int col0, int k0, int lane) {
  const int n = lane & 15, half = lane >> 4;
  FragU f;
#pragma unroll
  for (int r = 0; r < 8; ++r) {
    const int k = k0 + ((r >> 2) * 16) + half * 8 + 2 * (r & 3);
    f.u[r] = *(const unsigned int*)(base + (size_t)(col0 + n) * ld + k);
  }
  return f.v;
}

static __device__ __forceinline__ v8f wmma_f16(v16h a, v16h b, v8f c) {
  return __builtin_amdgcn_wmma_f32_16x16x32_f16(false, a, false, b, (short)0, c,
                                                false, false);
}

static __device__ __forceinline__ float redmax16(float v) {
#pragma unroll
  for (int m = 8; m >= 1; m >>= 1) v = fmaxf(v, __shfl_xor(v, m, 16));
  return v;
}
static __device__ __forceinline__ float redsum16(float v) {
#pragma unroll
  for (int m = 8; m >= 1; m >>= 1) v += __shfl_xor(v, m, 16);
  return v;
}

// ---- async 16B global->LDS copy (per-lane addresses), sync fallback ----
static __device__ __forceinline__ void cp16_to_lds(const _Float16* g, _Float16* l) {
#if USE_ASYNC
  __builtin_amdgcn_global_load_async_to_lds_b128(
      (__attribute__((address_space(1))) v4i*)(v4i*)(_Float16*)g,
      (__attribute__((address_space(3))) v4i*)(v4i*)l, 0, 0);
#else
  *(v4i*)l = *(const v4i*)g;
#endif
}
static __device__ __forceinline__ void async_wait_all() {
#if USE_ASYNC
  __builtin_amdgcn_s_wait_asynccnt(0);
#endif
}
// wait until only the most recent 16 async ops (the prefetch group) may remain
static __device__ __forceinline__ void async_wait_group() {
#if USE_ASYNC
  __builtin_amdgcn_s_wait_asynccnt(16);
#endif
}

// K tile: rows c0..c0+31 of kh[S,D] are one contiguous 4KB block. 8 issues/wave.
static __device__ __forceinline__ void stageK(const _Float16* kb, int c0,
                                              _Float16* dst, int lane) {
  const _Float16* src = kb + (size_t)c0 * DD;
#pragma unroll
  for (int i = 0; i < 8; ++i) {
    const int off = (i * 32 + lane) * 8;  // halves; 16B per lane-chunk
    cp16_to_lds(src + off, dst + off);
  }
}
// V tile: columns c0..c0+31 of vT[D,S] -> Vlds[64][32] f16. 8 issues/wave.
static __device__ __forceinline__ void stageV(const _Float16* vb, int c0,
                                              _Float16* dst, int lane) {
#pragma unroll
  for (int i = 0; i < 8; ++i) {
    const int lin = i * 32 + lane;       // 256 x 16B chunks
    const int d = lin >> 2, q = lin & 3;
    cp16_to_lds(vb + (size_t)d * SS + c0 + q * 8, dst + lin * 8);
  }
}

// ---------------- conversion kernels ----------------
__global__ void k_cast_f16(const float* __restrict__ in, _Float16* __restrict__ out, int n) {
  int i = blockIdx.x * blockDim.x + threadIdx.x;
  if (i < n) out[i] = (_Float16)in[i];
}

// out[c*R + r] = (f16) in[r*C + c]   (build Bt = W^T as [N,K] f16)
__global__ void k_transpose_cast(const float* __restrict__ in, _Float16* __restrict__ out,
                                 int R, int C) {
  int i = blockIdx.x * blockDim.x + threadIdx.x;
  if (i < R * C) {
    int r = i / C, c = i % C;
    out[(size_t)c * R + r] = (_Float16)in[i];
  }
}

// ------- WMMA GEMM: one wave -> 16x64 tile; A fragment reused across 4 WMMAs -------
__global__ __launch_bounds__(32) void k_gemm_wmma(const _Float16* __restrict__ A,
                                                  const _Float16* __restrict__ Bt,
                                                  float* __restrict__ C,
                                                  int M, int N, int K) {
  const int lane = threadIdx.x;
  const int row0 = blockIdx.x * 16;
  const int col0 = blockIdx.y * 64;
  v8f acc[4] = {v8f{}, v8f{}, v8f{}, v8f{}};
  for (int k0 = 0; k0 < K; k0 += 32) {
    const v16h a = load_fragA(A, K, row0, k0, lane);
#pragma unroll
    for (int t = 0; t < 4; ++t)
      acc[t] = wmma_f16(a, load_fragB(Bt, K, col0 + 16 * t, k0, lane), acc[t]);
  }
  const int n = lane & 15, half = lane >> 4;
#pragma unroll
  for (int t = 0; t < 4; ++t)
#pragma unroll
    for (int j = 0; j < 8; ++j)
      C[(size_t)(row0 + j + 8 * half) * N + col0 + 16 * t + n] = acc[t][j];
}

// ---------------- per-head RMSNorm + multidim RoPE ----------------
// One wave per (b,s,slot); slot 0..15 = q heads, 16..23 = k heads, 24..31 = v heads.
// Lane l owns channels l and l+32. cpd = 32, so rotate_half partner = lane ^ 16.
__global__ __launch_bounds__(32) void k_norm_rope(const float* __restrict__ qkv,
                                                  const float* __restrict__ cosb,
                                                  const float* __restrict__ sinb,
                                                  const float* __restrict__ qw,
                                                  const float* __restrict__ kw,
                                                  _Float16* __restrict__ qh,
                                                  _Float16* __restrict__ kh,
                                                  _Float16* __restrict__ vT) {
  const int lane = threadIdx.x;
  const int slot = blockIdx.x & 31;
  const int bs   = blockIdx.x >> 5;
  const int b = bs / SS, s = bs % SS;

  int col0;
  if (slot < NH)            col0 = slot * DD;
  else if (slot < NH + NKV) col0 = NH * DD + (slot - NH) * DD;
  else                      col0 = NH * DD + NKV * DD + (slot - NH - NKV) * DD;

  const float* row = qkv + (size_t)bs * NQKV + col0;
  float x0 = row[lane], x1 = row[lane + 32];
  float ssq = x0 * x0 + x1 * x1;
#pragma unroll
  for (int m = 16; m >= 1; m >>= 1) ssq += __shfl_xor(ssq, m, 32);
  const float rinv = rsqrtf(ssq * (1.0f / DD) + EPS);

  if (slot < NH + NKV) {  // q or k: scale + rope
    const float* w = (slot < NH) ? qw : kw;
    float y0 = x0 * rinv * w[lane];
    float y1 = x1 * rinv * w[lane + 32];
    float r0 = __shfl_xor(y0, 16, 32); r0 = (lane < 16) ? -r0 : r0;
    float r1 = __shfl_xor(y1, 16, 32); r1 = (lane < 16) ? -r1 : r1;
    const float* cb = cosb + (size_t)bs * DD;
    const float* sb = sinb + (size_t)bs * DD;
    float o0 = y0 * cb[lane]      + r0 * sb[lane];
    float o1 = y1 * cb[lane + 32] + r1 * sb[lane + 32];
    _Float16* dst;
    if (slot < NH) dst = qh + ((size_t)(b * NH + slot) * SS + s) * DD;
    else           dst = kh + ((size_t)(b * NKV + (slot - NH)) * SS + s) * DD;
    dst[lane]      = (_Float16)o0;
    dst[lane + 32] = (_Float16)o1;
  } else {                // v: norm only, store transposed [B,KV,D,S]
    const int kvh = slot - NH - NKV;
    _Float16* dst = vT + ((size_t)(b * NKV + kvh) * DD) * SS;
    dst[(size_t)lane * SS + s]        = (_Float16)(x0 * rinv);
    dst[(size_t)(lane + 32) * SS + s] = (_Float16)(x1 * rinv);
  }
}

// -------- flash attention: one wave per 16-row Q tile, async double-buffered K/V --------
__global__ __launch_bounds__(32) void k_flash_attn(const _Float16* __restrict__ qh,
                                                   const _Float16* __restrict__ kh,
                                                   const _Float16* __restrict__ vT,
                                                   _Float16* __restrict__ ctx) {
  __shared__ __align__(16) _Float16 Kl[2][32 * DD];   // 2 x 4KB
  __shared__ __align__(16) _Float16 Vl[2][DD * 32];   // 2 x 4KB
  __shared__ __align__(16) _Float16 Pl[16 * 32];      // 1KB
  const int lane = threadIdx.x;
  const int n = lane & 15, half = lane >> 4;
  const int qt = blockIdx.x, h = blockIdx.y, b = blockIdx.z;
  const int kv = h / (NH / NKV);

  const _Float16* qb = qh + ((size_t)(b * NH + h) * SS) * DD;
  const _Float16* kb = kh + ((size_t)(b * NKV + kv) * SS) * DD;
  const _Float16* vb = vT + ((size_t)(b * NKV + kv) * DD) * SS;

  const v16h qa0 = load_fragA(qb, DD, qt * 16, 0, lane);
  const v16h qa1 = load_fragA(qb, DD, qt * 16, 32, lane);

  v8f o[4] = {v8f{}, v8f{}, v8f{}, v8f{}};
  float Mr[8], Lr[8];
#pragma unroll
  for (int j = 0; j < 8; ++j) { Mr[j] = -1e30f; Lr[j] = 0.0f; }

  stageK(kb, 0, Kl[0], lane);       // 8 async issues
  stageV(vb, 0, Vl[0], lane);       // 8 async issues

  for (int c0 = 0; c0 < SS; c0 += 32) {
    const int cur = (c0 >> 5) & 1;
    if (c0 + 32 < SS) {             // prefetch next chunk into the other buffer
      stageK(kb, c0 + 32, Kl[cur ^ 1], lane);
      stageV(vb, c0 + 32, Vl[cur ^ 1], lane);
      async_wait_group();           // in-order: current chunk's 16 ops are done
    } else {
      async_wait_all();
    }
    __syncthreads();
    const _Float16* Kc = Kl[cur];
    const _Float16* Vc = Vl[cur];

    // scores for 32 keys: two 16x16 tiles, D=64 contraction => 2 WMMAs each
    v8f s0 = {}, s1 = {};
    s0 = wmma_f16(qa0, load_fragB(Kc, DD, 0, 0, lane), s0);
    s0 = wmma_f16(qa1, load_fragB(Kc, DD, 0, 32, lane), s0);
    s1 = wmma_f16(qa0, load_fragB(Kc, DD, 16, 0, lane), s1);
    s1 = wmma_f16(qa1, load_fragB(Kc, DD, 16, 32, lane), s1);

    // online softmax; C-tile row m = j + 8*half spans 16 lanes of one VGPR slot
#pragma unroll
    for (int j = 0; j < 8; ++j) {
      float cm = redmax16(fmaxf(s0[j], s1[j]));
      float nm = fmaxf(Mr[j], cm);
      float sc = __expf(Mr[j] - nm);
      Mr[j] = nm;
      float p0 = __expf(s0[j] - nm);
      float p1 = __expf(s1[j] - nm);
      Lr[j] = Lr[j] * sc + redsum16(p0 + p1);
      const int m = j + 8 * half;
      Pl[m * 32 + n]      = (_Float16)p0;
      Pl[m * 32 + 16 + n] = (_Float16)p1;
      o[0][j] *= sc; o[1][j] *= sc; o[2][j] *= sc; o[3][j] *= sc;
    }
    __syncthreads();

    // re-pack P (16x32) from LDS into A-fragment layout
    FragU pf;
#pragma unroll
    for (int r = 0; r < 8; ++r) {
      const int kk = ((r >> 2) * 16) + half * 8 + 2 * (r & 3);
      pf.u[r] = *(const unsigned int*)(&Pl[(lane & 15) * 32 + kk]);
    }
    const v16h pa = pf.v;

    // O += P(16x32) x V_chunk(32x64): 4 column blocks of 16
#pragma unroll
    for (int t = 0; t < 4; ++t)
      o[t] = wmma_f16(pa, load_fragB(Vc, 32, t * 16, 0, lane), o[t]);
    __syncthreads();
  }

#pragma unroll
  for (int j = 0; j < 8; ++j) {
    const float li = (Lr[j] > 0.0f) ? 1.0f / Lr[j] : 0.0f;
    const int row = qt * 16 + j + 8 * half;
#pragma unroll
    for (int t = 0; t < 4; ++t)
      ctx[((size_t)b * SS + row) * (NH * DD) + h * DD + t * 16 + n] =
          (_Float16)(o[t][j] * li);
  }
}

// ---------------- launcher ----------------
extern "C" void kernel_launch(void* const* d_in, const int* in_sizes, int n_in,
                              void* d_out, int out_size, void* d_ws, size_t ws_size,
                              hipStream_t stream) {
  (void)in_sizes; (void)n_in; (void)out_size; (void)ws_size;
  const float* hidden = (const float*)d_in[0];
  const float* cosb   = (const float*)d_in[1];
  const float* sinb   = (const float*)d_in[2];
  // d_in[3] = position_ids (int64): only its last-dim size (=2) matters; baked in.
  const float* Wq = (const float*)d_in[4];
  const float* Wk = (const float*)d_in[5];
  const float* Wv = (const float*)d_in[6];
  const float* Wo = (const float*)d_in[7];
  const float* qw = (const float*)d_in[8];
  const float* kw = (const float*)d_in[9];
  float* out = (float*)d_out;

  char* ws = (char*)d_ws;
  size_t off = 0;
  auto alloc = [&](size_t bytes) -> char* {
    char* p = ws + off;
    off += (bytes + 255) & ~(size_t)255;
    return p;
  };
  const int M = BB * SS;  // 4096
  _Float16* Xh    = (_Float16*)alloc((size_t)M * HID * 2);
  _Float16* Wqkvt = (_Float16*)alloc((size_t)NQKV * HID * 2);
  _Float16* Wot   = (_Float16*)alloc((size_t)HID * HID * 2);
  float*    qkv32 = (float*)alloc((size_t)M * NQKV * 4);
  _Float16* qh    = (_Float16*)alloc((size_t)BB * NH  * SS * DD * 2);
  _Float16* khp   = (_Float16*)alloc((size_t)BB * NKV * SS * DD * 2);
  _Float16* vT    = (_Float16*)alloc((size_t)BB * NKV * SS * DD * 2);
  _Float16* ctxH  = (_Float16*)qkv32;  // qkv32 is dead after norm/rope: reuse

  // 1) cast + transpose weights
  { int nel = M * HID;
    k_cast_f16<<<(nel + 255) / 256, 256, 0, stream>>>(hidden, Xh, nel); }
  { int nel = HID * (NH * DD);
    k_transpose_cast<<<(nel + 255) / 256, 256, 0, stream>>>(Wq, Wqkvt, HID, NH * DD); }
  { int nel = HID * (NKV * DD);
    k_transpose_cast<<<(nel + 255) / 256, 256, 0, stream>>>(
        Wk, Wqkvt + (size_t)(NH * DD) * HID, HID, NKV * DD); }
  { int nel = HID * (NKV * DD);
    k_transpose_cast<<<(nel + 255) / 256, 256, 0, stream>>>(
        Wv, Wqkvt + (size_t)(NH * DD + NKV * DD) * HID, HID, NKV * DD); }
  { int nel = (NH * DD) * HID;
    k_transpose_cast<<<(nel + 255) / 256, 256, 0, stream>>>(Wo, Wot, NH * DD, HID); }

  // 2) fused QKV projection (WMMA, 16x64 tile per wave)
  k_gemm_wmma<<<dim3(M / 16, NQKV / 64), 32, 0, stream>>>(Xh, Wqkvt, qkv32, M, NQKV, HID);

  // 3) per-head RMSNorm + RoPE, reshuffle to attention layouts
  k_norm_rope<<<M * 32, 32, 0, stream>>>(qkv32, cosb, sinb, qw, kw, qh, khp, vT);

  // 4) flash attention (WMMA QK^T + PV, async double-buffered K/V in LDS)
  k_flash_attn<<<dim3(SS / 16, NH, BB), 32, 0, stream>>>(qh, khp, vT, ctxH);

  // 5) output projection (WMMA), fp32 result straight to d_out
  k_gemm_wmma<<<dim3(M / 16, HID / 64), 32, 0, stream>>>(ctxH, Wot, out, M, HID, HID);
}
